// CustomLoss_science_guided_upweighing_7215545058015
// MI455X (gfx1250) — matrix operand order repaired and checked by reading, not compiled, not run on hardware
//
#include <hip/hip_runtime.h>
#include <hip/hip_bf16.h>
#include <math.h>

// ---------------------------------------------------------------------------
// CustomLoss (science-guided upweighing) for MI455X / gfx1250.
//   total = 0.5*mae + 0.1*fss + 0.1*prc + 0.3*landc ; outputs (total, mae)
// Memory-bound: ~256MB compulsory reads -> ~11us floor @ 23.3 TB/s; the FSS
// pass re-reads input+target (128MB, L2-resident after pass 1).
// FSS vertical 32-tap box sum: D = A_band(exact 0/1 f16) x H via
// v_wmma_f32_16x16x32_f16 (f32 accumulate). Halo tiles are copied with
// GLOBAL_LOAD_ASYNC_TO_LDS_B128 (ASYNCcnt) instead of VGPR round-trips.
// ---------------------------------------------------------------------------

typedef __attribute__((ext_vector_type(16))) _Float16 v16h;
typedef __attribute__((ext_vector_type(8)))  float    v8f;

#define IMG_H 1024
#define IMG_W 1024
#define NBATCH 16
#define NCLS 16
#define NPOS 993                      // 1024 - 32 + 1
#define NTOT (16.0 * 1024.0 * 1024.0) // elements per tensor

// workspace accumulator layout (doubles):
//  [0..15]  s1x per class   [16..31] s2x per class
//  [32..47] s1t per class   [48..63] s2t per class
//  [64] sum|x-t|   [65] sum(|x-t|*pb*mask)   [66] FSS comp count
__global__ void k_zero(double* acc) {
  int t = threadIdx.x;
  if (t < 72) acc[t] = 0.0;
}

// ---------------- elementwise reductions (streaming, float4) ---------------
__global__ __launch_bounds__(256) void k_elem(const float4* __restrict__ inp,
                                              const float4* __restrict__ tgt,
                                              const int4*   __restrict__ lc,
                                              const float4* __restrict__ pb,
                                              double* __restrict__ acc, int n4) {
  __shared__ float sCls[64];
  __shared__ float sA, sP;
  const int tid = threadIdx.x;
  if (tid < 64) sCls[tid] = 0.0f;
  if (tid == 0) { sA = 0.0f; sP = 0.0f; }
  __syncthreads();

  float la = 0.0f, lp = 0.0f;
  for (int i = blockIdx.x * 256 + tid; i < n4; i += gridDim.x * 256) {
    float4 t = tgt[i];
    float4 x = inp[i];
    float4 p = pb[i];
    int4   k = lc[i];
    float tv[4] = {t.x, t.y, t.z, t.w};
    float xv[4] = {x.x, x.y, x.z, x.w};
    float pv[4] = {p.x, p.y, p.z, p.w};
    int   kv[4] = {k.x, k.y, k.z, k.w};
#pragma unroll
    for (int j = 0; j < 4; ++j) {
      float msk = (tv[j] != -1.0f) ? 1.0f : 0.0f;
      float xm = xv[j] * msk;
      float tm = tv[j] * msk;
      float d  = fabsf(xm - tm);
      la += d;
      lp += d * pv[j] * msk;
      int cls = kv[j] & 15;
      atomicAdd(&sCls[cls],      xm);
      atomicAdd(&sCls[16 + cls], xm * xm);
      atomicAdd(&sCls[32 + cls], tm);
      atomicAdd(&sCls[48 + cls], tm * tm);
    }
  }
  atomicAdd(&sA, la);
  atomicAdd(&sP, lp);
  __syncthreads();
  if (tid < 64) atomicAdd(&acc[tid], (double)sCls[tid]);
  if (tid == 64) atomicAdd(&acc[64], (double)sA);
  if (tid == 65) atomicAdd(&acc[65], (double)sP);
}

// ------------------------------- FSS ---------------------------------------
// Block = 256 threads = 8 waves; block tile = 32x64 window positions; each
// wave computes one 16x16 tile. Halo region per block: 63 rows x 95 cols
// (row-padded to 96 = 24 b128 chunks). mask(target != -1) is applied in the
// horizontal pass so the halo can be copied raw with async b128 -> LDS.

// masked value: mask comes from T; value from T (useT) or X
__device__ __forceinline__ float fss_mval(const float (*__restrict__ X)[96],
                                          const float (*__restrict__ T)[96],
                                          int r, int c, bool useT) {
  float t = T[r][c];
  float v = useT ? t : X[r][c];
  return (t != -1.0f) ? v : 0.0f;
}

// horizontal 32-wide running sums: H[row][c] = sum_{j<32} mval(row, c+j)
__device__ __forceinline__ void fss_horiz(const float (*__restrict__ X)[96],
                                          const float (*__restrict__ T)[96],
                                          float (*__restrict__ H)[65],
                                          int tid, bool useT) {
  if (tid < 252) {
    int row = tid >> 2;
    int c0  = (tid & 3) * 16;
    float s = 0.0f;
#pragma unroll
    for (int j = 0; j < 32; ++j) s += fss_mval(X, T, row, c0 + j, useT);
    H[row][c0] = s;
    for (int c = c0 + 1; c < c0 + 16; ++c) {
      s += fss_mval(X, T, row, c + 31, useT) - fss_mval(X, T, row, c - 1, useT);
      H[row][c] = s;
    }
  }
}

// vertical 32-tap box sum via WMMA: D[m][n] = sum_k A[m][k] * H[tr+k][tc+n]
// B 32x16 f16 layout (same as A): element j -> K = j + 8*half (j<8), 16 + 8*half + (j-8) (j>=8)
__device__ __forceinline__ v8f fss_vert_wmma(const float (*__restrict__ H)[65],
                                             int tr, int tc, int half, int mrow,
                                             v16h a0, v16h a1) {
  v8f d = {};
  v16h bm;
#pragma unroll
  for (int j = 0; j < 16; ++j) {
    int k = ((j < 8) ? j : 8 + j) + 8 * half;    // chunk q=0: K 0..31
    bm[j] = (k < 47) ? (_Float16)H[tr + k][tc + mrow] : (_Float16)0.0f;
  }
  d = __builtin_amdgcn_wmma_f32_16x16x32_f16(false, a0, false, bm, (short)0, d, false, false);
#pragma unroll
  for (int j = 0; j < 16; ++j) {
    int k = 32 + ((j < 8) ? j : 8 + j) + 8 * half; // chunk q=1: K 32..63
    bm[j] = (k < 47) ? (_Float16)H[tr + k][tc + mrow] : (_Float16)0.0f;
  }
  d = __builtin_amdgcn_wmma_f32_16x16x32_f16(false, a1, false, bm, (short)0, d, false, false);
  return d;
}

__global__ __launch_bounds__(256) void k_fss(const float* __restrict__ inp,
                                             const float* __restrict__ tgt,
                                             double* __restrict__ acc) {
  __shared__ float RegX[63][96];   // raw input halo (stride 96: 24 b128/row)
  __shared__ float RegT[63][96];   // raw target halo
  __shared__ float Hs[63][65];     // horizontal sums, stride 65 (bank-safe)
  __shared__ float sComp;

  const int tid  = threadIdx.x;
  const int lane = tid & 31;
  const int wave = tid >> 5;
  const int tr = (wave >> 2) * 16;          // wave tile row offset in block
  const int tc = (wave & 3) * 16;           // wave tile col offset in block
  const int BR = blockIdx.y * 32;
  const int BC = blockIdx.x * 64;
  const size_t imgBase = (size_t)blockIdx.z * (size_t)(IMG_H * IMG_W);

  if (tid == 0) sComp = 0.0f;

  // Edge blocks: pre-zero the halo so out-of-image b128 chunks stay zero.
  if (blockIdx.x == 15 || blockIdx.y == 31) {
    float* fx = &RegX[0][0];
    float* ft = &RegT[0][0];
    for (int idx = tid; idx < 63 * 96; idx += 256) {
      fx[idx] = 0.0f;
      ft[idx] = 0.0f;
    }
  }
  __syncthreads();

  // Async halo copy: per-lane GLOBAL_LOAD_ASYNC_TO_LDS_B128 (ASYNCcnt).
  // 63 rows x 24 four-float chunks; chunks never straddle the image edge
  // (BC and IMG_W are multiples of 4), so predication handles OOB exactly.
  for (int idx = tid; idx < 63 * 24; idx += 256) {
    int i  = idx / 24;
    int jj = (idx - i * 24) * 4;
    int r = BR + i, c = BC + jj;
    if (r < IMG_H && c < IMG_W) {
      size_t off = imgBase + (size_t)r * IMG_W + (size_t)c;
      unsigned dx = (unsigned)(size_t)&RegX[i][jj];
      unsigned dt = (unsigned)(size_t)&RegT[i][jj];
      const float* gx = inp + off;
      const float* gt = tgt + off;
      asm volatile("global_load_async_to_lds_b128 %0, %1, off"
                   :: "v"(dx), "v"(gx) : "memory");
      asm volatile("global_load_async_to_lds_b128 %0, %1, off"
                   :: "v"(dt), "v"(gt) : "memory");
    }
  }
  asm volatile("s_wait_asynccnt 0x0" ::: "memory");
  __syncthreads();

  // Two 16x32 f16 band A-matrices (exact 0/1), K chunks q=0,1.
  // 16-bit A layout (ISA 7.12.2): klocal = (v&3)*2+i + (v>=4?16:0) + 8*half
  const int half = lane >> 4;
  const int mrow = lane & 15;
  v16h a0, a1;
#pragma unroll
  for (int v = 0; v < 8; ++v) {
#pragma unroll
    for (int i = 0; i < 2; ++i) {
      int klocal = ((v & 3) * 2 + i) + ((v >= 4) ? 16 : 0) + 8 * half;
      int k0 = klocal;
      int k1 = 32 + klocal;
      a0[v * 2 + i] = (k0 >= mrow && k0 < mrow + 32 && k0 < 47) ? (_Float16)1.0f : (_Float16)0.0f;
      a1[v * 2 + i] = (k1 >= mrow && k1 < mrow + 32 && k1 < 47) ? (_Float16)1.0f : (_Float16)0.0f;
    }
  }

  // ---- phase X: input*mask ----
  fss_horiz(RegX, RegT, Hs, tid, false);
  __syncthreads();
  v8f accX = fss_vert_wmma(Hs, tr, tc, half, mrow, a0, a1);
  __syncthreads();                 // Hs reused below

  // ---- phase T: target*mask ----
  fss_horiz(RegX, RegT, Hs, tid, true);
  __syncthreads();
  v8f accT = fss_vert_wmma(Hs, tr, tc, half, mrow, a0, a1);

  // classify every window position (C/D layout: M = v + 8*half, N = lane&15)
  const float T_DO_A = 0.2f * 1024.0f;   // 204.8
  const float T_OE_A = 0.4f * 1024.0f;   // 409.6
  float compSum = 0.0f;
#pragma unroll
  for (int v = 0; v < 8; ++v) {
    int m = v + 8 * half;
    int r = BR + tr + m;
    int c = BC + tc + mrow;
    float is = accX[v];
    float ts = accT[v];
    bool iw = is >= T_OE_A;
    bool io = (is >= T_DO_A) && (is < T_OE_A);
    bool id = is < T_DO_A;
    bool tw = ts >= T_OE_A;
    bool to = (ts >= T_DO_A) && (is < T_OE_A);   // reference's (intentional) bug kept
    bool td = ts < T_DO_A;
    bool match = (iw && tw) || (io && to) || (id && td);
    if (match && r < NPOS && c < NPOS) compSum += 1.0f;
  }
  atomicAdd(&sComp, compSum);
  __syncthreads();
  if (tid == 0) atomicAdd(&acc[66], (double)sComp);
}

// ------------------------------ finalize -----------------------------------
__global__ void k_fin(const double* __restrict__ acc, float* __restrict__ out) {
  if (threadIdx.x != 0 || blockIdx.x != 0) return;
  const double N = NTOT;
  double lcsum = 0.0;
  for (int k = 0; k < NCLS; ++k) {
    double vX = (acc[16 + k] - acc[k] * acc[k] / N) / (N - 1.0);
    double vT = (acc[48 + k] - acc[32 + k] * acc[32 + k] / N) / (N - 1.0);
    double sX = sqrt(vX > 0.0 ? vX : 0.0);
    double sT = sqrt(vT > 0.0 ? vT : 0.0);
    lcsum += (sT - sX);
  }
  double landc = lcsum / (double)NCLS;
  double mae = acc[64] / N;
  double prc = acc[65] / N;
  double fss = 1.0 - acc[66] / (1024.0 * (double)NBATCH * (double)NPOS * (double)NPOS);
  double total = 0.5 * mae + 0.1 * fss + 0.1 * prc + 0.3 * landc;
  out[0] = (float)total;
  out[1] = (float)mae;
}

// ---------------------------------------------------------------------------
extern "C" void kernel_launch(void* const* d_in, const int* in_sizes, int n_in,
                              void* d_out, int out_size, void* d_ws, size_t ws_size,
                              hipStream_t stream) {
  const float* inp = (const float*)d_in[0];
  const float* tgt = (const float*)d_in[1];
  const int*   lc  = (const int*)  d_in[2];
  const float* pb  = (const float*)d_in[3];
  double* acc = (double*)d_ws;
  float*  out = (float*)d_out;
  (void)in_sizes; (void)n_in; (void)out_size; (void)ws_size;

  k_zero<<<1, 128, 0, stream>>>(acc);

  const int n4 = (int)(16u * 1024u * 1024u / 4u);
  k_elem<<<4096, 256, 0, stream>>>((const float4*)inp, (const float4*)tgt,
                                   (const int4*)lc, (const float4*)pb, acc, n4);

  // FSS: 16 col-blocks x 32 row-blocks x 16 batches, 256 threads each
  k_fss<<<dim3(16, 32, 16), 256, 0, stream>>>(inp, tgt, acc);

  k_fin<<<1, 32, 0, stream>>>(acc, out);
}
